// SubstructureAwarePooling_75849122447747
// MI455X (gfx1250) — matrix operand order repaired and checked by reading, not compiled, not run on hardware
//
#include <hip/hip_runtime.h>

// ---------------- dims (fixed by the reference) ----------------
#define N_NODES 200000
#define D_DIM   256
#define S_SUB   8
#define M_IDX   100000
#define H_DIM   64
#define B_BATCH 4096

#define WAVES_PER_BLOCK 4
#define BLOCK_THREADS   (WAVES_PER_BLOCK * 32)
#define TILES_PER_WAVE  8
#define TILES_M         ((M_IDX + 15) / 16)                                   // 6250
#define TILES_PER_BLOCK (WAVES_PER_BLOCK * TILES_PER_WAVE)                    // 32
#define BLOCKS_PER_S    ((TILES_M + TILES_PER_BLOCK - 1) / TILES_PER_BLOCK)   // 196

// ---------------- CDNA5 WMMA vector types ----------------
typedef __attribute__((ext_vector_type(16))) __bf16 v16bf;
typedef __attribute__((ext_vector_type(8)))  __bf16 v8bf;
typedef __attribute__((ext_vector_type(4)))  __bf16 v4bf;
typedef __attribute__((ext_vector_type(8)))  float  v8f;

// ---------------- zero-init of output ----------------
__global__ void sap_zero_kernel(float* __restrict__ out, int n) {
    int i = blockIdx.x * blockDim.x + threadIdx.x;
    if (i < n) out[i] = 0.0f;
}

// ---------------- one-pass node-feature f32 -> bf16 (coalesced) ----------------
// in: [N*D] f32, out: [N*D] bf16.  Each thread converts 4 elements.
__global__ void sap_cvt_nodes_kernel(const float* __restrict__ in,
                                     unsigned short* __restrict__ out, int n4) {
    int i = blockIdx.x * blockDim.x + threadIdx.x;
    if (i < n4) {
        float4 v = ((const float4*)in)[i];
        v4bf o;
        o[0] = (__bf16)v.x; o[1] = (__bf16)v.y;
        o[2] = (__bf16)v.z; o[3] = (__bf16)v.w;
        ((v4bf*)out)[i] = o;
    }
}

// ---------------- fused gather -> MLP(WMMA bf16) -> scatter-add ----------------
// grid = (BLOCKS_PER_S, S_SUB), block = 128 threads (4 waves),
// each wave handles TILES_PER_WAVE consecutive 16-row tiles.
template <bool PRECONV>
__global__ void __launch_bounds__(BLOCK_THREADS)
sap_main_kernel(const float* __restrict__ nodef,            // [N, D] f32
                const unsigned short* __restrict__ nodebf,  // [N, D] bf16 (if PRECONV)
                const float* __restrict__ W1,               // [S, D, H]
                const float* __restrict__ b1,               // [S, H]
                const float* __restrict__ W2,               // [S, H, H]
                const float* __restrict__ b2,               // [S, H]
                const int*   __restrict__ subidx,           // [S, M]
                const int*   __restrict__ bidx,             // [N]
                float*       __restrict__ out)              // [B, S*H]
{
    // W1 fragments: [kc=8][nt=4][lane=32][j=16] bf16  -> 32 KB
    __shared__ __bf16 w1f[8 * 4 * 32 * 16];
    // W2 fragments: [kc=2][nt=4][lane=32][j=16] bf16  ->  8 KB
    __shared__ __bf16 w2f[2 * 4 * 32 * 16];
    // per-wave h staging: 16 rows x 64 cols bf16      ->  8 KB total
    __shared__ __bf16 hbuf[WAVES_PER_BLOCK][16 * H_DIM];

    const int s   = blockIdx.y;
    const int tid = threadIdx.x;

    // ---- cooperatively convert W1[s], W2[s] to bf16 in B-fragment order ----
    // B-fragment (16-bit, KxN=32x16): element j of lane L is
    //   K = kc*32 + (j<8 ? 8*(L>>4)+j : 16+8*(L>>4)+(j-8)),  N = nt*16 + (L&15)
    const float* W1s = W1 + (size_t)s * D_DIM * H_DIM;
    const float* W2s = W2 + (size_t)s * H_DIM * H_DIM;
    for (int e = tid; e < 8 * 4 * 32 * 16; e += BLOCK_THREADS) {
        int j    = e & 15;
        int lane = (e >> 4) & 31;
        int nt   = (e >> 9) & 3;
        int kc   = e >> 11;
        int k = kc * 32 + ((j & 8) << 1) + ((lane >> 4) << 3) + (j & 7);
        int n = nt * 16 + (lane & 15);
        w1f[e] = (__bf16)W1s[k * H_DIM + n];
    }
    for (int e = tid; e < 2 * 4 * 32 * 16; e += BLOCK_THREADS) {
        int j    = e & 15;
        int lane = (e >> 4) & 31;
        int nt   = (e >> 9) & 3;
        int kc   = e >> 11;
        int k = kc * 32 + ((j & 8) << 1) + ((lane >> 4) << 3) + (j & 7);
        int n = nt * 16 + (lane & 15);
        w2f[e] = (__bf16)W2s[k * H_DIM + n];
    }
    __syncthreads();

    const int wave  = tid >> 5;
    const int lane  = tid & 31;
    const int row   = lane & 15;          // M row handled by this lane (A layout)
    const int kb    = (lane >> 4) << 3;   // 0 or 8: per-lane-half K sub-base
    const int ncol  = lane & 15;          // N column handled by this lane (B & C layouts)
    const int rbase = (lane >= 16) ? 8 : 0;

    // per-lane bias values (invariant over tiles)
    float b1v[4], b2v[4];
#pragma unroll
    for (int nt = 0; nt < 4; ++nt) {
        b1v[nt] = b1[s * H_DIM + nt * 16 + ncol];
        b2v[nt] = b2[s * H_DIM + nt * 16 + ncol];
    }

    __bf16* hrow = &hbuf[wave][0];
    const int tile0 = (blockIdx.x * WAVES_PER_BLOCK + wave) * TILES_PER_WAVE;

#pragma unroll 1
    for (int t = 0; t < TILES_PER_WAVE; ++t) {
        const int tile = tile0 + t;
        const int m0   = tile * 16;
        const bool active = (m0 < M_IDX);

        // gather: this lane's node row (clamped so tail waves stay in-bounds)
        int m = m0 + row;
        if (m >= M_IDX) m = M_IDX - 1;
        const int node = subidx[s * M_IDX + m];

        // ================= layer 1: [16x256] x [256x64] =================
        v8f acc[4] = {};
        if (PRECONV) {
            const __bf16* arow = (const __bf16*)nodebf + (size_t)node * D_DIM;
#pragma unroll
            for (int kc = 0; kc < 8; ++kc) {
                // A fragment (16-bit, MxK=16x32): j=0..7 -> K=kc*32+kb+j,
                //                                 j=8..15 -> K=kc*32+16+kb+(j-8)
                v8bf a0 = *(const v8bf*)(arow + kc * 32 + kb);
                v8bf a1 = *(const v8bf*)(arow + kc * 32 + 16 + kb);
                v16bf a = __builtin_shufflevector(a0, a1, 0, 1, 2, 3, 4, 5, 6, 7,
                                                  8, 9, 10, 11, 12, 13, 14, 15);
#pragma unroll
                for (int nt = 0; nt < 4; ++nt) {
                    v16bf bf = *(const v16bf*)&w1f[((kc * 4 + nt) * 32 + lane) * 16];
                    acc[nt] = __builtin_amdgcn_wmma_f32_16x16x32_bf16(
                        false, a, false, bf, (short)0, acc[nt], false, false);
                }
            }
        } else {
            const float* arow = nodef + (size_t)node * D_DIM;
#pragma unroll
            for (int kc = 0; kc < 8; ++kc) {
                const float4* p0 = (const float4*)(arow + kc * 32 + kb);
                const float4* p1 = (const float4*)(arow + kc * 32 + 16 + kb);
                float4 g0a = p0[0], g0b = p0[1];
                float4 g1a = p1[0], g1b = p1[1];
                v16bf a;
                a[0]  = (__bf16)g0a.x; a[1]  = (__bf16)g0a.y;
                a[2]  = (__bf16)g0a.z; a[3]  = (__bf16)g0a.w;
                a[4]  = (__bf16)g0b.x; a[5]  = (__bf16)g0b.y;
                a[6]  = (__bf16)g0b.z; a[7]  = (__bf16)g0b.w;
                a[8]  = (__bf16)g1a.x; a[9]  = (__bf16)g1a.y;
                a[10] = (__bf16)g1a.z; a[11] = (__bf16)g1a.w;
                a[12] = (__bf16)g1b.x; a[13] = (__bf16)g1b.y;
                a[14] = (__bf16)g1b.z; a[15] = (__bf16)g1b.w;
#pragma unroll
                for (int nt = 0; nt < 4; ++nt) {
                    v16bf bf = *(const v16bf*)&w1f[((kc * 4 + nt) * 32 + lane) * 16];
                    acc[nt] = __builtin_amdgcn_wmma_f32_16x16x32_bf16(
                        false, a, false, bf, (short)0, acc[nt], false, false);
                }
            }
        }

        // ---- bias + ReLU, stage h tile (16x64 bf16) in this wave's LDS slice ----
        // C/D layout: element r of lane L is (M = r + rbase, N = ncol + 16*nt)
#pragma unroll
        for (int nt = 0; nt < 4; ++nt) {
#pragma unroll
            for (int r = 0; r < 8; ++r) {
                float h = acc[nt][r] + b1v[nt];
                h = (h > 0.0f) ? h : 0.0f;
                hrow[(rbase + r) * H_DIM + nt * 16 + ncol] = (__bf16)h;
            }
        }
        // wave-private LDS region: RAW within this wave only -> counter wait suffices
        asm volatile("s_wait_dscnt 0" ::: "memory");

        // ================= layer 2: [16x64] x [64x64] =================
        v8f acc2[4] = {};
#pragma unroll
        for (int kc = 0; kc < 2; ++kc) {
            const __bf16* hp = hrow + row * H_DIM + kc * 32 + kb;
            v8bf a0 = *(const v8bf*)hp;          // K = kc*32 + kb + 0..7
            v8bf a1 = *(const v8bf*)(hp + 16);   // K = kc*32 + 16 + kb + 0..7
            v16bf a = __builtin_shufflevector(a0, a1, 0, 1, 2, 3, 4, 5, 6, 7,
                                              8, 9, 10, 11, 12, 13, 14, 15);
#pragma unroll
            for (int nt = 0; nt < 4; ++nt) {
                v16bf bf = *(const v16bf*)&w2f[((kc * 4 + nt) * 32 + lane) * 16];
                acc2[nt] = __builtin_amdgcn_wmma_f32_16x16x32_bf16(
                    false, a, false, bf, (short)0, acc2[nt], false, false);
            }
        }

        // ================= bias + segment scatter-add =================
        if (active) {
#pragma unroll
            for (int r = 0; r < 8; ++r) {
                int mm = m0 + rbase + r;
                if (mm < M_IDX) {
                    int nd = subidx[s * M_IDX + mm];
                    int bslot = bidx[nd];
                    float* orow = out + (size_t)bslot * (S_SUB * H_DIM) + s * H_DIM;
#pragma unroll
                    for (int nt = 0; nt < 4; ++nt) {
                        atomicAdd(&orow[nt * 16 + ncol], acc2[nt][r] + b2v[nt]);
                    }
                }
            }
        }
    }
}

extern "C" void kernel_launch(void* const* d_in, const int* in_sizes, int n_in,
                              void* d_out, int out_size, void* d_ws, size_t ws_size,
                              hipStream_t stream) {
    const float* nodef  = (const float*)d_in[0];
    const float* W1     = (const float*)d_in[1];
    const float* b1     = (const float*)d_in[2];
    const float* W2     = (const float*)d_in[3];
    const float* b2     = (const float*)d_in[4];
    const int*   subidx = (const int*)d_in[5];
    const int*   bidx   = (const int*)d_in[6];
    float*       out    = (float*)d_out;

    // zero output (harness poisons it; scatter is atomic accumulate)
    int zb = (out_size + 255) / 256;
    sap_zero_kernel<<<zb, 256, 0, stream>>>(out, out_size);

    dim3 grid(BLOCKS_PER_S, S_SUB);
    const size_t need = (size_t)N_NODES * D_DIM * sizeof(unsigned short);
    if (ws_size >= need) {
        // pre-convert node features to bf16 (102 MB -> fully L2-resident gather table)
        unsigned short* nodebf = (unsigned short*)d_ws;
        int n4 = N_NODES * D_DIM / 4;
        sap_cvt_nodes_kernel<<<(n4 + 255) / 256, 256, 0, stream>>>(nodef, nodebf, n4);
        sap_main_kernel<true><<<grid, BLOCK_THREADS, 0, stream>>>(
            nodef, nodebf, W1, b1, W2, b2, subidx, bidx, out);
    } else {
        sap_main_kernel<false><<<grid, BLOCK_THREADS, 0, stream>>>(
            nodef, nullptr, W1, b1, W2, b2, subidx, bidx, out);
    }
}
